// DualRoadRevAttnGNN_28965259444643
// MI455X (gfx1250) — matrix operand order repaired and checked by reading, not compiled, not run on hardware
//
#include <hip/hip_runtime.h>
#include <hip/hip_bf16.h>

// ---------------------------------------------------------------------------
// DualRoadRevAttnGNN for MI455X (gfx1250): WMMA bf16 GEMMs + sparse GCN agg.
// Round 2: b128 LDS fragment loads, software-pipelined B fragments,
//          wave-level shuffle softmax, vectorized bf16 staging.
// ---------------------------------------------------------------------------

#define C_DIM 128
#define NPG   128

typedef __attribute__((ext_vector_type(16))) __bf16 v16bf;
typedef __attribute__((ext_vector_type(8)))  float  v8f;

union FragU { uint4 q[2]; unsigned int u[8]; v16bf v; };

__device__ __forceinline__ unsigned short f2bf(float f) {
  unsigned int u = __float_as_uint(f);
  unsigned int r = u + 0x7FFFu + ((u >> 16) & 1u);   // round-to-nearest-even
  return (unsigned short)(r >> 16);
}

// A-matrix 16x32 bf16 fragment (ISA 7.12.2).  Per lane the 16 halfwords are two
// contiguous 16B runs: [half*8 .. +7] and [16+half*8 .. +7]  ->  2x ds_load_b128.
__device__ __forceinline__ v16bf load_a_frag(const unsigned short* base, int m0,
                                             int ldk, int k0) {
  const int lane = threadIdx.x & 31;
  const unsigned short* p = base + (m0 + (lane & 15)) * ldk + k0 + (lane >> 4) * 8;
  FragU f;
  f.q[0] = *(const uint4*)(p);
  f.q[1] = *(const uint4*)(p + 16);
  return f.v;
}

// B-matrix 32x16 bf16 fragment from n-major tile Bt[n][k]: per lane two
// contiguous 16B runs: [half*8 .. +7] and [16+half*8 .. +7]  ->  2x ds_load_b128.
__device__ __forceinline__ v16bf load_b_frag(const unsigned short* base, int n0,
                                             int ldk, int k0) {
  const int lane = threadIdx.x & 31;
  const unsigned short* p = base + (n0 + (lane & 15)) * ldk + k0 + (lane >> 4) * 8;
  FragU f;
  f.q[0] = *(const uint4*)(p);
  f.q[1] = *(const uint4*)(p + 16);
  return f.v;
}

// 16x128 output strip accumulation over K=128: software-pipelined so the next
// B fragment's ds_load_b128 pair is in flight while the current WMMA executes.
__device__ __forceinline__ void wmma_strip_128(const unsigned short* Amat,
                                               const unsigned short* Bmat,
                                               int mloc, v8f acc[8]) {
#pragma unroll
  for (int ks = 0; ks < 4; ++ks) {
    v16bf af = load_a_frag(Amat, mloc, 128, ks * 32);
    v16bf bf = load_b_frag(Bmat, 0, 128, ks * 32);
#pragma unroll
    for (int nt = 0; nt < 8; ++nt) {
      v16bf bn = bf;
      if (nt < 7) bn = load_b_frag(Bmat, (nt + 1) * 16, 128, ks * 32);
      acc[nt] = __builtin_amdgcn_wmma_f32_16x16x32_bf16(
          false, af, false, bf, (short)0, acc[nt], false, false);
      bf = bn;
    }
  }
}

// Stage a 128x128 fp32 tile into LDS as bf16, same layout (float4 -> b64 store).
__device__ __forceinline__ void stage_rowmajor_bf16(unsigned short* dst,
                                                    const float* src, int tid) {
  for (int idx = tid; idx < 128 * 128 / 4; idx += 256) {
    float4 v = *(const float4*)(src + (size_t)idx * 4);
    ushort4 o;
    o.x = f2bf(v.x); o.y = f2bf(v.y); o.z = f2bf(v.z); o.w = f2bf(v.w);
    *(ushort4*)(dst + idx * 4) = o;
  }
}

// Stage a 128x128 fp32 tile transposed into LDS as bf16 (coalesced reads).
__device__ __forceinline__ void stage_transposed_bf16(unsigned short* dst,
                                                      const float* src, int tid) {
  for (int idx = tid; idx < 128 * 128; idx += 256) {
    int r = idx >> 7, c = idx & 127;
    dst[c * 128 + r] = f2bf(src[idx]);
  }
}

// ---------------------------------------------------------------------------
// Generic GEMM: out[M,128] = act( concat-A[M,128*nchunks] @ W[128*nchunks,128]
//                                 + bias ).  mode: 0 identity, 1 sigmoid.
// Block = 256 threads (8 wave32); 64 KB dynamic LDS.
// ---------------------------------------------------------------------------
__global__ __launch_bounds__(256) void gemm_wmma_bf16(
    const float* __restrict__ A0, const float* __restrict__ A1,
    const float* __restrict__ W, const float* __restrict__ bias,
    float* __restrict__ out, int nchunks, int mode) {
  extern __shared__ char smem[];
  unsigned short* Ws = (unsigned short*)smem;                   // [n][k]
  unsigned short* As = (unsigned short*)(smem + 128 * 128 * 2); // [m][k]

  const int tid  = threadIdx.x;
  const int wave = tid >> 5;
  const int lane = tid & 31;
  const int rowBase = blockIdx.x * 128;
  const int mloc = wave * 16;
  const int nl   = lane & 15;
  const int half = lane >> 4;

  v8f acc[8];
#pragma unroll
  for (int nt = 0; nt < 8; ++nt) {
    float b = bias ? bias[nt * 16 + nl] : 0.0f;
#pragma unroll
    for (int j = 0; j < 8; ++j) acc[nt][j] = b;
  }

  for (int kc = 0; kc < nchunks; ++kc) {
    const float* A  = (kc == 0) ? A0 : A1;
    const float* Wc = W + (size_t)kc * 128 * 128;
    stage_transposed_bf16(Ws, Wc, tid);
    stage_rowmajor_bf16(As, A + (size_t)rowBase * 128, tid);
    __syncthreads();
    wmma_strip_128(As, Ws, mloc, acc);
    __syncthreads();
  }

#pragma unroll
  for (int nt = 0; nt < 8; ++nt) {
    int n = nt * 16 + nl;
#pragma unroll
    for (int v = 0; v < 8; ++v) {
      int m = v + half * 8;
      float val = acc[nt][v];
      if (mode == 1) val = 1.0f / (1.0f + __expf(-val));
      out[(size_t)(rowBase + mloc + m) * 128 + n] = val;
    }
  }
}

// ---------------------------------------------------------------------------
// Attention stage 1: per-graph S = (Q K^T) / sqrt(C).  One block per graph.
// ---------------------------------------------------------------------------
__global__ __launch_bounds__(256) void attn_scores_wmma(
    const float* __restrict__ Q, const float* __restrict__ K,
    float* __restrict__ S) {
  extern __shared__ char smem[];
  unsigned short* Qs = (unsigned short*)smem;            // [q][d]  (A)
  unsigned short* Ks = (unsigned short*)(smem + 32768);  // [n][d]  (B, n-major)
  const size_t base = (size_t)blockIdx.x * 128 * 128;
  const int tid = threadIdx.x, wave = tid >> 5, lane = tid & 31;
  const int mloc = wave * 16, nl = lane & 15, half = lane >> 4;

  stage_rowmajor_bf16(Qs, Q + base, tid);
  stage_rowmajor_bf16(Ks, K + base, tid);
  __syncthreads();

  v8f acc[8];
#pragma unroll
  for (int nt = 0; nt < 8; ++nt)
#pragma unroll
    for (int j = 0; j < 8; ++j) acc[nt][j] = 0.0f;

  wmma_strip_128(Qs, Ks, mloc, acc);

  const float scale = 0.088388347648318447f;  // 1/sqrt(128)
#pragma unroll
  for (int nt = 0; nt < 8; ++nt)
#pragma unroll
    for (int v = 0; v < 8; ++v)
      S[base + (size_t)(mloc + v + half * 8) * 128 + nt * 16 + nl] = acc[nt][v] * scale;
}

// ---------------------------------------------------------------------------
// Attention stage 2: Pc = 0.5*(softmax(S) + softmax(-S)) per 128-wide row.
// One wave32 per row, 4 elements per lane, shuffle reductions (no barriers).
// ---------------------------------------------------------------------------
__global__ __launch_bounds__(256) void attn_dual_softmax(float* __restrict__ S,
                                                         int nrows) {
  const int w    = (blockIdx.x * blockDim.x + threadIdx.x) >> 5;
  const int lane = threadIdx.x & 31;
  if (w >= nrows) return;
  float* row = S + (size_t)w * 128 + lane * 4;
  float4 v = *(const float4*)row;

  float mx = fmaxf(fmaxf(v.x, v.y), fmaxf(v.z, v.w));
  float mn = fminf(fminf(v.x, v.y), fminf(v.z, v.w));
#pragma unroll
  for (int o = 16; o > 0; o >>= 1) {
    mx = fmaxf(mx, __shfl_xor(mx, o));
    mn = fminf(mn, __shfl_xor(mn, o));
  }
  float4 ep, en;
  ep.x = __expf(v.x - mx); ep.y = __expf(v.y - mx);
  ep.z = __expf(v.z - mx); ep.w = __expf(v.w - mx);
  en.x = __expf(mn - v.x); en.y = __expf(mn - v.y);
  en.z = __expf(mn - v.z); en.w = __expf(mn - v.w);
  float sp = ep.x + ep.y + ep.z + ep.w;
  float sn = en.x + en.y + en.z + en.w;
#pragma unroll
  for (int o = 16; o > 0; o >>= 1) {
    sp += __shfl_xor(sp, o);
    sn += __shfl_xor(sn, o);
  }
  float rp = 0.5f / sp, rn = 0.5f / sn;
  float4 o4;
  o4.x = ep.x * rp + en.x * rn; o4.y = ep.y * rp + en.y * rn;
  o4.z = ep.z * rp + en.z * rn; o4.w = ep.w * rp + en.w * rn;
  *(float4*)row = o4;
}

// ---------------------------------------------------------------------------
// Attention stage 3: per-graph attn = Pc @ V (V staged transposed: Vt[d][k]).
// ---------------------------------------------------------------------------
__global__ __launch_bounds__(256) void attn_pv_wmma(
    const float* __restrict__ Pc, const float* __restrict__ V,
    float* __restrict__ attn) {
  extern __shared__ char smem[];
  unsigned short* Ps = (unsigned short*)smem;            // [q][k]  (A)
  unsigned short* Vt = (unsigned short*)(smem + 32768);  // [d][k]  (B, n-major)
  const size_t base = (size_t)blockIdx.x * 128 * 128;
  const int tid = threadIdx.x, wave = tid >> 5, lane = tid & 31;
  const int mloc = wave * 16, nl = lane & 15, half = lane >> 4;

  stage_rowmajor_bf16(Ps, Pc + base, tid);
  stage_transposed_bf16(Vt, V + base, tid);
  __syncthreads();

  v8f acc[8];
#pragma unroll
  for (int nt = 0; nt < 8; ++nt)
#pragma unroll
    for (int j = 0; j < 8; ++j) acc[nt][j] = 0.0f;

  wmma_strip_128(Ps, Vt, mloc, acc);

#pragma unroll
  for (int nt = 0; nt < 8; ++nt)
#pragma unroll
    for (int v = 0; v < 8; ++v)
      attn[base + (size_t)(mloc + v + half * 8) * 128 + nt * 16 + nl] = acc[nt][v];
}

// ---------------------------------------------------------------------------
// GCN aggregation helpers
// ---------------------------------------------------------------------------
__global__ void deg_init_k(float* __restrict__ deg, int n) {
  int i = blockIdx.x * blockDim.x + threadIdx.x;
  if (i < n) deg[i] = 1.0f;                       // self-loop
}
__global__ void deg_accum_k(const int* __restrict__ ei, float* __restrict__ deg, int E) {
  int i = blockIdx.x * blockDim.x + threadIdx.x;
  if (i < E) atomicAdd(&deg[ei[E + i]], 1.0f);    // dst row
}
__global__ void deg_inv_k(float* __restrict__ deg, int n) {
  int i = blockIdx.x * blockDim.x + threadIdx.x;
  if (i < n) deg[i] = rsqrtf(fmaxf(deg[i], 1.0f));
}
// t[i,c] = conv_b[c] + dinv[i]^2 * t0[i,c]   (self-loop contribution + bias)
__global__ void gcn_selfinit_k(const float* __restrict__ t0, const float* __restrict__ dinv,
                               const float* __restrict__ bias, float* __restrict__ t,
                               int total) {
  int i = blockIdx.x * blockDim.x + threadIdx.x;
  if (i < total) {
    int node = i >> 7, c = i & 127;
    float dv = dinv[node];
    t[i] = bias[c] + dv * dv * t0[i];
  }
}
// one wave per edge, 4 channels per lane
__global__ __launch_bounds__(256) void gcn_scatter_k(
    const float* __restrict__ t0, const int* __restrict__ ei,
    const float* __restrict__ dinv, float* __restrict__ t, int E) {
  int e = (blockIdx.x * blockDim.x + threadIdx.x) >> 5;
  int lane = threadIdx.x & 31;
  if (e >= E) return;
  int s = ei[e], d = ei[E + e];
  float w = dinv[s] * dinv[d];
  const float4 v = *(const float4*)(t0 + (size_t)s * 128 + lane * 4);
  float* o = t + (size_t)d * 128 + lane * 4;
  atomicAdd(o + 0, w * v.x);
  atomicAdd(o + 1, w * v.y);
  atomicAdd(o + 2, w * v.z);
  atomicAdd(o + 3, w * v.w);
}

// GraphNorm + LeakyReLU, in place.  Block per graph, thread per channel.
__global__ __launch_bounds__(128) void graphnorm_k(
    float* __restrict__ t, const float* __restrict__ w, const float* __restrict__ b,
    const float* __restrict__ ms, float eps) {
  const int g = blockIdx.x, c = threadIdx.x;
  const size_t base = (size_t)g * NPG * 128 + c;
  float sum = 0.0f;
  for (int n = 0; n < NPG; ++n) sum += t[base + (size_t)n * 128];
  float mean = sum * (1.0f / NPG);
  float mc = ms[c] * mean;
  float var = 0.0f;
  for (int n = 0; n < NPG; ++n) { float d = t[base + (size_t)n * 128] - mc; var += d * d; }
  var *= (1.0f / NPG);
  float sc = w[c] * rsqrtf(var + eps);
  float bb = b[c];
  for (int n = 0; n < NPG; ++n) {
    float v = (t[base + (size_t)n * 128] - mc) * sc + bb;
    t[base + (size_t)n * 128] = (v >= 0.0f) ? v : 0.01f * v;   // leaky_relu
  }
}

// h = gate*t + (1-gate)*rev + prev
__global__ void fuse_res_k(const float* __restrict__ gate, const float* __restrict__ t,
                           const float* __restrict__ rev, const float* __restrict__ prev,
                           float* __restrict__ h, int total) {
  int i = blockIdx.x * blockDim.x + threadIdx.x;
  if (i < total) {
    float g = gate[i];
    h[i] = g * t[i] + (1.0f - g) * rev[i] + prev[i];
  }
}

// gf = pooled(h1) + 2*pooled(h2)   (reference's all_x[i-1] quirk), counts=128
__global__ __launch_bounds__(128) void pool_k(const float* __restrict__ h1,
                                              const float* __restrict__ h2,
                                              float* __restrict__ gf) {
  const int g = blockIdx.x, c = threadIdx.x;
  const size_t base = (size_t)g * NPG * 128 + c;
  float s1 = 0.0f, s2 = 0.0f;
  for (int n = 0; n < NPG; ++n) {
    s1 += h1[base + (size_t)n * 128];
    s2 += h2[base + (size_t)n * 128];
  }
  gf[(size_t)g * 128 + c] = (s1 + 2.0f * s2) * (1.0f / NPG);
}

__global__ void tail_zero_k(float* __restrict__ out, int from, int size) {
  int i = from + blockIdx.x * blockDim.x + threadIdx.x;
  if (i < size) out[i] = 0.0f;
}

// ---------------------------------------------------------------------------
extern "C" void kernel_launch(void* const* d_in, const int* in_sizes, int n_in,
                              void* d_out, int out_size, void* d_ws, size_t ws_size,
                              hipStream_t stream) {
  const float* x      = (const float*)d_in[0];
  const int*   ei     = (const int*)d_in[1];
  /* batch (d_in[2]) unused: contiguous equal segments of 128 */
  const float* emb_W  = (const float*)d_in[3];
  const float* emb_b  = (const float*)d_in[4];
  const float* conv_W = (const float*)d_in[5];
  const float* conv_b = (const float*)d_in[6];
  const float* gn_w   = (const float*)d_in[7];
  const float* gn_b   = (const float*)d_in[8];
  const float* gn_ms  = (const float*)d_in[9];
  const float* qW = (const float*)d_in[10]; const float* qb = (const float*)d_in[11];
  const float* kW = (const float*)d_in[12]; const float* kb = (const float*)d_in[13];
  const float* vW = (const float*)d_in[14]; const float* vb = (const float*)d_in[15];
  const float* oW = (const float*)d_in[16]; const float* ob = (const float*)d_in[17];
  const float* fW = (const float*)d_in[18]; const float* fb = (const float*)d_in[19];

  const int N = in_sizes[0] / C_DIM;        // 32768
  const int E = in_sizes[1] / 2;            // 524288
  const int G = N / NPG;                    // 256
  const size_t NC = (size_t)N * C_DIM;

  float* ws  = (float*)d_ws;
  float* hA  = ws;                 // layer-0 input (embedding output)
  float* h1  = hA + NC;            // layer-0 output
  float* h2  = h1 + NC;            // layer-1 output
  float* t0  = h2 + NC;            // conv pre-agg; later aliased by S / attn
  float* t   = t0 + NC;            // post-agg / post-norm activations
  float* Qb  = t + NC;             // Q; later aliased by rev
  float* Kb  = Qb + NC;            // K; later aliased by gate
  float* Vb  = Kb + NC;            // V
  float* dinv = Vb + NC;           // degree^-1/2 [N]
  float* Sb   = t0;                // scores / Pc / attn (t0 dead post-aggregation)
  float* rev  = Qb;
  float* gate = Kb;

  const dim3 blk256(256), blk128(128);
  const int gemmGrid = N / 128;             // 256
  const size_t gemmLds = 2 * 128 * 128 * 2; // 64 KB

  // degree -> dinv (in place)
  deg_init_k<<<dim3((N + 255) / 256), blk256, 0, stream>>>(dinv, N);
  deg_accum_k<<<dim3((E + 255) / 256), blk256, 0, stream>>>(ei, dinv, E);
  deg_inv_k<<<dim3((N + 255) / 256), blk256, 0, stream>>>(dinv, N);

  // embedding: hA = x @ emb_W + emb_b
  gemm_wmma_bf16<<<dim3(gemmGrid), blk256, gemmLds, stream>>>(
      x, nullptr, emb_W, emb_b, hA, 1, 0);

  const float* hin = hA;
  float* houts[2] = {h1, h2};
  for (int i = 0; i < 2; ++i) {
    // t0 = hin @ conv_W[i]  (bias added post-aggregation per reference)
    gemm_wmma_bf16<<<dim3(gemmGrid), blk256, gemmLds, stream>>>(
        hin, nullptr, conv_W + (size_t)i * C_DIM * C_DIM, nullptr, t0, 1, 0);
    // t = conv_b + dinv^2*t0 (+ edge scatter)
    gcn_selfinit_k<<<dim3((int)(NC / 256)), blk256, 0, stream>>>(
        t0, dinv, conv_b + i * C_DIM, t, (int)NC);
    gcn_scatter_k<<<dim3(E / 8), blk256, 0, stream>>>(t0, ei, dinv, t, E);
    // GraphNorm + LeakyReLU (in place on t)
    graphnorm_k<<<dim3(G), blk128, 0, stream>>>(
        t, gn_w + i * C_DIM, gn_b + i * C_DIM, gn_ms + i * C_DIM, 1e-5f);
    // Q/K/V projections
    gemm_wmma_bf16<<<dim3(gemmGrid), blk256, gemmLds, stream>>>(t, nullptr, qW, qb, Qb, 1, 0);
    gemm_wmma_bf16<<<dim3(gemmGrid), blk256, gemmLds, stream>>>(t, nullptr, kW, kb, Kb, 1, 0);
    gemm_wmma_bf16<<<dim3(gemmGrid), blk256, gemmLds, stream>>>(t, nullptr, vW, vb, Vb, 1, 0);
    // attention: S = QK^T*scale ; Pc = 0.5*(softmax(S)+softmax(-S)) ; attn = Pc@V
    attn_scores_wmma<<<dim3(G), blk256, gemmLds, stream>>>(Qb, Kb, Sb);
    attn_dual_softmax<<<dim3(N / 8), blk256, 0, stream>>>(Sb, N);
    attn_pv_wmma<<<dim3(G), blk256, gemmLds, stream>>>(Sb, Vb, Sb);
    // rev = attn @ oW + ob   (Qb dead -> rev)
    gemm_wmma_bf16<<<dim3(gemmGrid), blk256, gemmLds, stream>>>(Sb, nullptr, oW, ob, rev, 1, 0);
    // gate = sigmoid(concat(t, rev) @ fW + fb)   (Kb dead -> gate)
    gemm_wmma_bf16<<<dim3(gemmGrid), blk256, gemmLds, stream>>>(t, rev, fW, fb, gate, 2, 1);
    // h = gate*t + (1-gate)*rev + hin
    fuse_res_k<<<dim3((int)(NC / 256)), blk256, 0, stream>>>(
        gate, t, rev, hin, houts[i], (int)NC);
    hin = houts[i];
  }

  // gf = pooled(h1) + 2*pooled(h2)
  pool_k<<<dim3(G), blk128, 0, stream>>>(h1, h2, (float*)d_out);
  int gc = G * C_DIM;
  if (out_size > gc)
    tail_zero_k<<<dim3((out_size - gc + 255) / 256), blk256, 0, stream>>>(
        (float*)d_out, gc, out_size);
}
